// GNNAutoencoder_52158082842633
// MI455X (gfx1250) — matrix operand order repaired and checked by reading, not compiled, not run on hardware
//
#include <hip/hip_runtime.h>
#include <hip/hip_bf16.h>

typedef __attribute__((ext_vector_type(16))) _Float16 v16h;
typedef __attribute__((ext_vector_type(8)))  float    v8f;
typedef __attribute__((ext_vector_type(4)))  unsigned u32x4;
typedef __attribute__((ext_vector_type(8)))  int      i32x8;
typedef __attribute__((ext_vector_type(4)))  int      i32x4;

#define IN_DIM 128
#define HID    256
#define LAT    64

// Hardware float atomic add (no return -> STOREcnt path, global_atomic_add_f32)
__device__ __forceinline__ void atomAddF(float* p, float v) {
    __hip_atomic_fetch_add(p, v, __ATOMIC_RELAXED, __HIP_MEMORY_SCOPE_AGENT);
}

// ---------------- GCN normalization ----------------
__global__ __launch_bounds__(256) void k_deg_init(float* deg, int n) {
    int i = blockIdx.x * blockDim.x + threadIdx.x;
    if (i < n) deg[i] = 1.0f;   // self-loop weight 1
}

__global__ __launch_bounds__(256) void k_deg_accum(const long long* __restrict__ col,
                                                   const float* __restrict__ ew,
                                                   float* deg, int e) {
    int i = blockIdx.x * blockDim.x + threadIdx.x;
    if (i < e) atomAddF(&deg[col[i]], ew[i]);
}

__global__ __launch_bounds__(256) void k_dinv(float* deg, int n) {
    int i = blockIdx.x * blockDim.x + threadIdx.x;
    if (i < n) { float d = deg[i]; deg[i] = (d > 0.0f) ? rsqrtf(d) : 0.0f; }
}

__global__ __launch_bounds__(256) void k_norm(const long long* __restrict__ row,
                                              const long long* __restrict__ col,
                                              const float* __restrict__ ew,
                                              const float* __restrict__ dinv,
                                              float* __restrict__ nrm, int e) {
    int i = blockIdx.x * blockDim.x + threadIdx.x;
    if (i < e) nrm[i] = dinv[row[i]] * ew[i] * dinv[col[i]];
}

// ---------------- weight pre-pass: W[K,Nout] f32 -> Wt[Nout,K] f16 (transposed) ----------------
__global__ __launch_bounds__(256)
void k_cvtW(const float* __restrict__ W, _Float16* __restrict__ Wt,
            int total, int K, int nShift, int nMask)
{
    int idx = blockIdx.x * blockDim.x + threadIdx.x;
    if (idx >= total) return;
    int k = idx >> nShift;
    int j = idx & nMask;
    Wt[(size_t)j * K + k] = (_Float16)W[idx];
}

// ---------------- WMMA GEMM: Out[M,Nout] = act(A[M,K]) @ W[K,Nout] ----------------
// Block = 8 waves, covers 8 M-tiles x 64 output columns.
// Weight panel Wt[n0..n0+63][0..K-1] (f16) staged into LDS by ONE Tensor Data
// Mover descriptor per block: 2D tile (tile_dim0=K, tile_dim1=64) with the TDM
// LDS-padding feature inserting 8 dwords (16 halves) after every K halves, so
// LDS layout is [col][K+16] -> each B fragment is a contiguous 32B-aligned run
// read with ds_load_b128. Each wave computes 4 adjacent 16x16 N-tiles: one A
// fragment feeds 4 v_wmma_f32_16x16x32_f16; K-loop fully unrolled.
template<int K, bool RELU>
__global__ __launch_bounds__(256)
void k_gemm_wmma(const float* __restrict__ A, const _Float16* __restrict__ Wt,
                 float* __restrict__ Out, int M, int Nout)
{
    constexpr int LDK = K + 16;               // padded stride (halves); mult. of 16
    __shared__ _Float16 ldsW[64 * LDK];

    const int n0   = blockIdx.y << 6;         // 64-column panel
    const int tid  = threadIdx.x;
    const int wave = tid >> 5;

    if (wave == 0) {                          // one TDM issue per block
        unsigned long long ga = (unsigned long long)(const void*)(Wt + (size_t)n0 * K);
        unsigned lds = (unsigned)(unsigned long long)(const void*)&ldsW[0];

        // ---- D# group 0: count=1 | lds_addr | global_addr(57b) | type=2
        u32x4 g0;
        g0[0] = 1u;                                               // count=1, user mode
        g0[1] = lds;                                              // LDS byte address
        g0[2] = (unsigned)(ga & 0xFFFFFFFFull);                   // global_addr[31:0]
        g0[3] = (unsigned)((ga >> 32) & 0x1FFFFFFull) | (2u << 30); // [56:32] | type=2

        // ---- D# group 1: data_size=2B, pad after every K halves by 16 halves
        constexpr unsigned padInterval = (K == 256) ? 6u : (K == 128) ? 5u : 4u; // 2^(v+1) dwords = K/2
        i32x8 g1;
        g1[0] = (int)((1u << 16) | (1u << 20) | (padInterval << 22) | (7u << 25));
        g1[1] = (int)(((unsigned)K & 0xFFFFu) << 16);             // abar=0 | tensor_dim0 lo
        g1[2] = (int)((((unsigned)Nout & 0xFFFFu) << 16));        // td0 hi=0 | tensor_dim1 lo
        g1[3] = (int)((unsigned)K << 16);                         // td1 hi=0 | tile_dim0=K
        g1[4] = 64;                                               // tile_dim1=64 | tile_dim2=0
        g1[5] = K;                                                // tensor_dim0_stride[31:0]=K
        g1[6] = 0;                                                // stride0 hi | stride1 lo
        g1[7] = 0;                                                // stride1 hi
        i32x4 gz = {0, 0, 0, 0};
#if defined(__clang_major__) && (__clang_major__ >= 23)
        i32x8 gz8 = {0, 0, 0, 0, 0, 0, 0, 0};
        __builtin_amdgcn_tensor_load_to_lds(g0, g1, gz, gz, gz8, 0);
#else
        __builtin_amdgcn_tensor_load_to_lds(g0, g1, gz, gz, 0);
#endif
        __builtin_amdgcn_s_wait_tensorcnt(0);
    }
    __syncthreads();

    const int lane   = tid & 31;
    const int mtile  = blockIdx.x * 8 + wave;
    const int mtiles = M >> 4;
    if (mtile < mtiles) {                     // wave-uniform: EXEC all-1s inside
        const int m0   = mtile << 4;
        const int half = lane >> 4;           // 0: lanes 0-15, 1: lanes 16-31
        const int l    = lane & 15;
        const float* __restrict__ arow = A + (size_t)(m0 + l) * K;

        v8f acc0 = {}, acc1 = {}, acc2 = {}, acc3 = {};
        #pragma unroll
        for (int k0 = 0; k0 < K; k0 += 32) {
            __builtin_prefetch(arow + k0 + 32, 0, 1);   // speculative; OOB dropped

            // ---- A fragment: halves[0..15]=K k0+half*16..k0+half*16+15
            const float4* apLo = (const float4*)(arow + k0 + half * 16);
            const float4* apHi = (const float4*)(arow + k0 + half * 16 + 8);
            float4 f0 = apLo[0], f1 = apLo[1];
            float4 f2 = apHi[0], f3 = apHi[1];
            if (RELU) {
                f0.x = fmaxf(f0.x, 0.f); f0.y = fmaxf(f0.y, 0.f);
                f0.z = fmaxf(f0.z, 0.f); f0.w = fmaxf(f0.w, 0.f);
                f1.x = fmaxf(f1.x, 0.f); f1.y = fmaxf(f1.y, 0.f);
                f1.z = fmaxf(f1.z, 0.f); f1.w = fmaxf(f1.w, 0.f);
                f2.x = fmaxf(f2.x, 0.f); f2.y = fmaxf(f2.y, 0.f);
                f2.z = fmaxf(f2.z, 0.f); f2.w = fmaxf(f2.w, 0.f);
                f3.x = fmaxf(f3.x, 0.f); f3.y = fmaxf(f3.y, 0.f);
                f3.z = fmaxf(f3.z, 0.f); f3.w = fmaxf(f3.w, 0.f);
            }
            v16h a;
            a[0]  = (_Float16)f0.x; a[1]  = (_Float16)f0.y;
            a[2]  = (_Float16)f0.z; a[3]  = (_Float16)f0.w;
            a[4]  = (_Float16)f1.x; a[5]  = (_Float16)f1.y;
            a[6]  = (_Float16)f1.z; a[7]  = (_Float16)f1.w;
            a[8]  = (_Float16)f2.x; a[9]  = (_Float16)f2.y;
            a[10] = (_Float16)f2.z; a[11] = (_Float16)f2.w;
            a[12] = (_Float16)f3.x; a[13] = (_Float16)f3.y;
            a[14] = (_Float16)f3.z; a[15] = (_Float16)f3.w;

            // ---- B fragments from LDS: col = t*16 + l, K = k0 + half*16 .. +15
            const int kB = k0 + half * 16;    // multiple of 16 -> 32B aligned
            const _Float16* bb = ldsW + kB + l * LDK;
            v16h b0 = *(const v16h*)(bb);
            v16h b1 = *(const v16h*)(bb + 16 * LDK);
            v16h b2 = *(const v16h*)(bb + 32 * LDK);
            v16h b3 = *(const v16h*)(bb + 48 * LDK);

            acc0 = __builtin_amdgcn_wmma_f32_16x16x32_f16(false, a, false, b0,
                                                          (short)0, acc0, false, false);
            acc1 = __builtin_amdgcn_wmma_f32_16x16x32_f16(false, a, false, b1,
                                                          (short)0, acc1, false, false);
            acc2 = __builtin_amdgcn_wmma_f32_16x16x32_f16(false, a, false, b2,
                                                          (short)0, acc2, false, false);
            acc3 = __builtin_amdgcn_wmma_f32_16x16x32_f16(false, a, false, b3,
                                                          (short)0, acc3, false, false);
        }

        // ---- D layout: VGPR r -> row m0 + half*8 + r, col n0 + t*16 + l
        float* __restrict__ orow = Out + (size_t)(m0 + half * 8) * Nout + (n0 + l);
        #pragma unroll
        for (int r = 0; r < 8; ++r) {
            float* p = orow + (size_t)r * Nout;
            p[0]  = acc0[r];
            p[16] = acc1[r];
            p[32] = acc2[r];
            p[48] = acc3[r];
        }
    }
}

// ---------------- aggregation: out = bias + dinv[i]^2 * g  (self-loop folded in) ----------------
__global__ __launch_bounds__(256)
void k_agg_init(const float* __restrict__ g, const float* __restrict__ dinv,
                const float* __restrict__ bias, float* __restrict__ out,
                int total, int fShift, int fMask)
{
    int idx = blockIdx.x * blockDim.x + threadIdx.x;
    if (idx >= total) return;
    int i = idx >> fShift;
    int f = idx & fMask;
    float dv = dinv[i];
    out[idx] = bias[f] + dv * dv * g[idx];
}

// ---------------- aggregation: scatter-add over edges, 4 floats per thread ----------------
__global__ __launch_bounds__(256)
void k_agg_scatter(const float* __restrict__ g, const long long* __restrict__ row,
                   const long long* __restrict__ col, const float* __restrict__ nrm,
                   float* __restrict__ out, long long total, int fvShift, int F)
{
    long long idx = (long long)blockIdx.x * blockDim.x + threadIdx.x;
    if (idx >= total) return;
    long long e = idx >> fvShift;                 // edge
    int c = (int)(idx & ((1 << fvShift) - 1));    // float4 chunk within feature row
    long long r  = row[e];
    long long cl = col[e];
    float nv = nrm[e];
    const float4 hv = *(const float4*)(g + r * (long long)F + c * 4);
    float* op = out + cl * (long long)F + c * 4;
    atomAddF(op + 0, hv.x * nv);
    atomAddF(op + 1, hv.y * nv);
    atomAddF(op + 2, hv.z * nv);
    atomAddF(op + 3, hv.w * nv);
}

static inline int cdiv(long long a, int b) { return (int)((a + b - 1) / b); }

extern "C" void kernel_launch(void* const* d_in, const int* in_sizes, int n_in,
                              void* d_out, int out_size, void* d_ws, size_t ws_size,
                              hipStream_t stream) {
    const float*     x   = (const float*)d_in[0];
    const long long* ei  = (const long long*)d_in[1];
    const float*     ew  = (const float*)d_in[2];
    const float*     W1  = (const float*)d_in[3];
    const float*     b1  = (const float*)d_in[4];
    const float*     W2  = (const float*)d_in[5];
    const float*     b2  = (const float*)d_in[6];
    const float*     W3  = (const float*)d_in[7];
    const float*     b3  = (const float*)d_in[8];
    const float*     W4  = (const float*)d_in[9];
    const float*     b4  = (const float*)d_in[10];

    const int n = in_sizes[0] / IN_DIM;   // 50000
    const int e = in_sizes[1] / 2;        // 800000
    const long long* row = ei;            // edge_index[0]
    const long long* col = ei + e;        // edge_index[1]

    float*     dinv  = (float*)d_ws;                       // [n]
    float*     nrm   = dinv + n;                           // [e]
    float*     bufA  = nrm + e;                            // [n*HID]  GEMM output
    float*     bufB  = bufA + (size_t)n * HID;             // [n*HID]  aggregated output
    _Float16*  whBuf = (_Float16*)(bufB + (size_t)n * HID);// [<=HID*HID] f16 transposed W
    float*     outp  = (float*)d_out;                      // [n*IN_DIM]

    // ---- normalization
    k_deg_init <<<cdiv(n, 256), 256, 0, stream>>>(dinv, n);
    k_deg_accum<<<cdiv(e, 256), 256, 0, stream>>>(col, ew, dinv, e);
    k_dinv     <<<cdiv(n, 256), 256, 0, stream>>>(dinv, n);
    k_norm     <<<cdiv(e, 256), 256, 0, stream>>>(row, col, ew, dinv, nrm, e);

    const int mtiles = n >> 4;  // 3125
    const int gx = cdiv(mtiles, 8);

    // ---- per-layer: cvt W -> f16^T, WMMA GEMM (TDM-staged), init(bias+self), edge scatter
    struct Agg { const float* b; int Nout; float* gbuf; float* obuf; };
    auto run_agg = [&](const Agg& L) {
        int fShift = (L.Nout == 256) ? 8 : (L.Nout == 128) ? 7 : 6;
        int total  = n * L.Nout;
        k_agg_init<<<cdiv(total, 256), 256, 0, stream>>>(L.gbuf, dinv, L.b, L.obuf,
                                                         total, fShift, L.Nout - 1);
        int fvShift = fShift - 2;
        long long stotal = (long long)e << fvShift;
        k_agg_scatter<<<cdiv(stotal, 256), 256, 0, stream>>>(L.gbuf, row, col, nrm,
                                                             L.obuf, stotal, fvShift, L.Nout);
    };

    // conv1: x[*,128] @ W1[128,256], no input relu
    k_cvtW<<<cdiv(IN_DIM * HID, 256), 256, 0, stream>>>(W1, whBuf, IN_DIM * HID, IN_DIM, 8, HID - 1);
    k_gemm_wmma<IN_DIM, false><<<dim3(gx, HID >> 6), 256, 0, stream>>>(x, whBuf, bufA, n, HID);
    run_agg({b1, HID, bufA, bufB});

    // conv2: relu(h)[*,256] @ W2[256,64]
    k_cvtW<<<cdiv(HID * LAT, 256), 256, 0, stream>>>(W2, whBuf, HID * LAT, HID, 6, LAT - 1);
    k_gemm_wmma<HID, true><<<dim3(gx, LAT >> 6), 256, 0, stream>>>(bufB, whBuf, bufA, n, LAT);
    run_agg({b2, LAT, bufA, bufB});

    // conv3: z[*,64] @ W3[64,256], no input relu
    k_cvtW<<<cdiv(LAT * HID, 256), 256, 0, stream>>>(W3, whBuf, LAT * HID, LAT, 8, HID - 1);
    k_gemm_wmma<LAT, false><<<dim3(gx, HID >> 6), 256, 0, stream>>>(bufB, whBuf, bufA, n, HID);
    run_agg({b3, HID, bufA, bufB});

    // conv4: relu(h)[*,256] @ W4[256,128] -> d_out
    k_cvtW<<<cdiv(HID * IN_DIM, 256), 256, 0, stream>>>(W4, whBuf, HID * IN_DIM, HID, 7, IN_DIM - 1);
    k_gemm_wmma<HID, true><<<dim3(gx, IN_DIM >> 6), 256, 0, stream>>>(bufB, whBuf, bufA, n, IN_DIM);
    run_agg({b4, IN_DIM, bufA, outp});
}
